// TVADecoder_44968307589758
// MI455X (gfx1250) — matrix operand order repaired and checked by reading, not compiled, not run on hardware
//
#include <hip/hip_runtime.h>
#include <cstdint>
#include <cstddef>

// ---------------- dims ----------------
#define NB   120
#define NTT  20
#define NL   144
#define ND   64
#define NH   1024
#define NA   256
#define NR   9216          // L*D
#define N4H  4096
#define MPAD 128
#define BL   (NB*NL)       // 17280, multiple of 16 (and 144 = 9*16)

// ---------------- vector types ----------------
typedef __bf16 v16bf __attribute__((ext_vector_type(16)));
typedef __bf16 v8bf  __attribute__((ext_vector_type(8)));
typedef float  v8f   __attribute__((ext_vector_type(8)));
typedef float  v4f   __attribute__((ext_vector_type(4)));

union BF16Frag { v16bf v; v8bf h[2]; };

// ---- fast transcendentals: use HW v_tanh_f32 / v_exp_f32 when available ----
static __device__ __forceinline__ float fast_tanh(float x) {
#if __has_builtin(__builtin_amdgcn_tanhf)
    return __builtin_amdgcn_tanhf(x);
#elif __has_builtin(__builtin_amdgcn_tanh_f32)
    return __builtin_amdgcn_tanh_f32(x);
#else
    return tanhf(x);
#endif
}
static __device__ __forceinline__ float fast_exp(float x) {
    return __expf(x);      // native v_exp_f32 path
}
static __device__ __forceinline__ float sigf(float x) {
    return 1.0f / (1.0f + fast_exp(-x));
}

static __device__ __forceinline__ __bf16 f2bf(float x) {
    union { float f; unsigned u; } in; in.f = x;
    unsigned r = in.u + 0x7FFFu + ((in.u >> 16) & 1u);   // round-to-nearest-even
    union { unsigned short s; __bf16 b; } out; out.s = (unsigned short)(r >> 16);
    return out.b;
}
static __device__ __forceinline__ __bf16 bfzero() {
    union { unsigned short s; __bf16 b; } out; out.s = 0; return out.b;
}

static __device__ __forceinline__ v8f wmma_bf16(const BF16Frag& a, const BF16Frag& b, v8f c) {
    // D = A(16x32 bf16) * B(32x16 bf16) + C(16x16 f32)
    return __builtin_amdgcn_wmma_f32_16x16x32_bf16(false, a.v, false, b.v,
                                                   (short)0, c, false, false);
}

// convert 8 consecutive floats -> v8bf
static __device__ __forceinline__ v8bf cvt8(const float* __restrict__ p) {
    v4f x = *(const v4f*)p;
    v4f y = *(const v4f*)(p + 4);
    v8bf r;
    r[0]=f2bf(x[0]); r[1]=f2bf(x[1]); r[2]=f2bf(x[2]); r[3]=f2bf(x[3]);
    r[4]=f2bf(y[0]); r[5]=f2bf(y[1]); r[6]=f2bf(y[2]); r[7]=f2bf(y[3]);
    return r;
}

// ============ one-time kernels ============

__global__ void __launch_bounds__(256) k_convert_bf16(const float* __restrict__ src,
                                                      __bf16* __restrict__ dst, int n) {
    int i = blockIdx.x * 256 + threadIdx.x;
    if (i < n) dst[i] = f2bf(src[i]);
}

// Wxa [D,A] (row d major) -> WxaT bf16 [A,D]
__global__ void __launch_bounds__(256) k_transpose_wxa(const float* __restrict__ Wxa,
                                                       __bf16* __restrict__ WxaT) {
    int i = blockIdx.x * 256 + threadIdx.x;     // i = a*64 + d
    if (i < NA * ND) {
        int a = i >> 6, d = i & 63;
        WxaT[i] = f2bf(Wxa[d * NA + a]);
    }
}

// zero padding rows 120..127 of v_bf [128,9216] and h_bf [128,1024]
__global__ void __launch_bounds__(256) k_zero_pad(__bf16* __restrict__ vbf,
                                                  __bf16* __restrict__ hbf) {
    int i = blockIdx.x * 256 + threadIdx.x;
    const int nv = 8 * NR;                      // 73728
    if (i < nv) vbf[NB * NR + i] = bfzero();
    else if (i < nv + 8 * NH) hbf[NB * NH + (i - nv)] = bfzero();
}

// f0[b,d] = mean_l feature[(b*T+0), l, d]
__global__ void __launch_bounds__(64) k_f0(const float* __restrict__ feature,
                                           float* __restrict__ f0) {
    int b = blockIdx.x, d = threadIdx.x;
    const float* base = feature + (size_t)(b * NTT) * NR + d;
    float s = 0.f;
    for (int l = 0; l < NL; ++l) s += base[l * ND];
    f0[b * ND + d] = s * (1.0f / NL);
}

// h0 = tanh(f0 @ Wh_w^T + Wh_b), c0 = tanh(f0 @ Wc_w^T + Wc_b)
__global__ void __launch_bounds__(256) k_hc0(const float* __restrict__ f0,
                                             const float* __restrict__ Whw, const float* __restrict__ Whb,
                                             const float* __restrict__ Wcw, const float* __restrict__ Wcb,
                                             float* __restrict__ hbuf, float* __restrict__ cbuf,
                                             __bf16* __restrict__ hbf) {
    int idx = blockIdx.x * 256 + threadIdx.x;   // b*1024 + j
    int b = idx >> 10, j = idx & 1023;
    const float* f = f0 + b * ND;
    const float* wh = Whw + (size_t)j * ND;
    const float* wc = Wcw + (size_t)j * ND;
    float hs = Whb[j], cs = Wcb[j];
    #pragma unroll 8
    for (int d = 0; d < ND; ++d) { float fv = f[d]; hs += fv * wh[d]; cs += fv * wc[d]; }
    float h = fast_tanh(hs), c = fast_tanh(cs);
    hbuf[idx] = h; cbuf[idx] = c; hbf[(size_t)b * NH + j] = f2bf(h);
}

// ============ per-step kernels ============

// ha[b,a] = h@Wha ; out_bet[(b*T+t),d] = sigmoid(h@Wb + bb)
__global__ void __launch_bounds__(320) k_attn_small(const float* __restrict__ hbuf,
                                                    const float* __restrict__ Wha,
                                                    const float* __restrict__ Wb,
                                                    const float* __restrict__ bb,
                                                    float* __restrict__ ha,
                                                    float* __restrict__ out_bet, int t) {
    int b = blockIdx.x, tid = threadIdx.x;
    const float* h = hbuf + (size_t)b * NH;
    if (tid < NA) {
        float s = 0.f;
        for (int j = 0; j < NH; ++j) s += h[j] * Wha[(size_t)j * NA + tid];
        ha[b * NA + tid] = s;
    } else {
        int d = tid - NA;                       // d < 64
        float s = bb[d];
        for (int j = 0; j < NH; ++j) s += h[j] * Wb[(size_t)j * ND + d];
        out_bet[(size_t)(b * NTT + t) * ND + d] = sigf(s);
    }
}

// e[b,l] = sum_a tanh((xt@Wxa)[b,l,a] + ha[b,a]) * wa[a]     (WMMA, A built on the fly)
__global__ void __launch_bounds__(128) k_score(const float* __restrict__ feature,
                                               const __bf16* __restrict__ WxaT, // [A=256, D=64]
                                               const float* __restrict__ ha,
                                               const float* __restrict__ wa,
                                               float* __restrict__ e, int t) {
    const int lane  = threadIdx.x & 31;
    const int wave  = threadIdx.x >> 5;
    const int mTile = blockIdx.x * 4 + wave;            // 1080 tiles
    const int rowA  = mTile * 16 + (lane & 15);         // global row in [0, 17280)
    const int bA    = rowA / NL, lA = rowA % NL;
    const int bTile = (mTile * 16) / NL;                // whole tile shares one b (144 = 9*16)
    const float* fr = feature + ((size_t)(bA * NTT + t) * NL + lA) * ND;
    const int aKoff = (lane >> 4) * 8;
    const int bKoff = (lane >> 4) * 8;
    const int colB  = lane & 15;

    BF16Frag a0, a1;                                    // K = 0..31 and 32..63
    a0.h[0] = cvt8(fr + 0  + aKoff); a0.h[1] = cvt8(fr + 16 + aKoff);
    a1.h[0] = cvt8(fr + 32 + aKoff); a1.h[1] = cvt8(fr + 48 + aKoff);

    float epart[8] = {0.f,0.f,0.f,0.f,0.f,0.f,0.f,0.f};

    for (int nt = 0; nt < 16; ++nt) {
        const int col = nt * 16 + colB;                 // attn dim index a
        const __bf16* wp = WxaT + (size_t)col * ND + bKoff;
        BF16Frag b0, b1;
        b0.h[0] = *(const v8bf*)(wp);       b0.h[1] = *(const v8bf*)(wp + 16);
        b1.h[0] = *(const v8bf*)(wp + 32);  b1.h[1] = *(const v8bf*)(wp + 48);
        v8f acc = {};
        acc = wmma_bf16(a0, b0, acc);
        acc = wmma_bf16(a1, b1, acc);
        float wav = wa[col];
        const float* hap = ha + bTile * NA;
        #pragma unroll
        for (int r = 0; r < 8; ++r)
            epart[r] += fast_tanh(acc[r] + hap[col]) * wav;
    }
    // reduce across 16 lanes within each half-wave (each half owns 8 rows)
    #pragma unroll
    for (int r = 0; r < 8; ++r) {
        float s = epart[r];
        for (int off = 8; off >= 1; off >>= 1) s += __shfl_xor(s, off, 16);
        if ((lane & 15) == 0)
            e[mTile * 16 + ((lane >> 4) << 3) + r] = s;
    }
}

// alpha = softmax_L(e); visual = xt*(alpha+1)*beta; writes vis/alpha outputs + bf16 v
__global__ void __launch_bounds__(256) k_softmax_visual(const float* __restrict__ feature,
                                                        const float* __restrict__ e,
                                                        const float* __restrict__ out_bet,
                                                        float* __restrict__ out_vis,
                                                        float* __restrict__ out_alp,
                                                        __bf16* __restrict__ vbf, int t) {
    __shared__ float sAlpha[NL];
    __shared__ float sred[8];
    const int b = blockIdx.x, tid = threadIdx.x, lane = tid & 31, wv = tid >> 5;

    float v = (tid < NL) ? e[b * NL + tid] : -3.0e38f;
    float m = v;
    for (int off = 16; off >= 1; off >>= 1) m = fmaxf(m, __shfl_xor(m, off, 32));
    if (lane == 0) sred[wv] = m;
    __syncthreads();
    float mx = sred[0];
    #pragma unroll
    for (int i = 1; i < 8; ++i) mx = fmaxf(mx, sred[i]);
    __syncthreads();

    float ex = (tid < NL) ? fast_exp(v - mx) : 0.f;
    float s = ex;
    for (int off = 16; off >= 1; off >>= 1) s += __shfl_xor(s, off, 32);
    if (lane == 0) sred[wv] = s;
    __syncthreads();
    float tot = 0.f;
    #pragma unroll
    for (int i = 0; i < 8; ++i) tot += sred[i];
    float alpha = ex / tot;
    if (tid < NL) {
        sAlpha[tid] = alpha;
        out_alp[(size_t)(b * NTT + t) * NL + tid] = alpha;
    }
    __syncthreads();

    const float* fb  = feature + (size_t)(b * NTT + t) * NR;
    float*       ov  = out_vis + (size_t)(b * NTT + t) * NR;
    __bf16*      vb  = vbf + (size_t)b * NR;
    const float* bet = out_bet + (size_t)(b * NTT + t) * ND;
    for (int idx = tid; idx < NR; idx += 256) {
        int l = idx >> 6, d = idx & 63;
        float val = fb[idx] * (sAlpha[l] + 1.0f) * bet[d];
        ov[idx] = val;
        vb[idx] = f2bf(val);
    }
}

// gates[128,4096] = v_bf[128,9216] @ Wih^T + h_bf[128,1024] @ Whh^T    (WMMA)
__global__ void __launch_bounds__(256) k_gates_gemm(const __bf16* __restrict__ Abf,  // [128,9216]
                                                    const __bf16* __restrict__ Hbf,  // [128,1024]
                                                    const __bf16* __restrict__ Wih,  // [4096,9216]
                                                    const __bf16* __restrict__ Whh,  // [4096,1024]
                                                    float* __restrict__ gates) {
    const int lane  = threadIdx.x & 31;
    const int wave  = threadIdx.x >> 5;                 // 8 waves -> 8 M-tiles (M=128)
    const int nBase = blockIdx.x * 128;                 // grid.x = 32 (N=4096)
    const int mBase = wave * 16;
    const int rowA  = mBase + (lane & 15);
    const int aKoff = (lane >> 4) * 8;
    const int bKoff = (lane >> 4) * 8;
    const int colB  = lane & 15;

    v8f acc[8] = {};

    // K over v (9216)
    for (int k = 0; k < NR; k += 32) {
        BF16Frag a;
        a.h[0] = *(const v8bf*)(Abf + (size_t)rowA * NR + k + aKoff);
        a.h[1] = *(const v8bf*)(Abf + (size_t)rowA * NR + k + 16 + aKoff);
        #pragma unroll
        for (int nt = 0; nt < 8; ++nt) {
            const __bf16* wp = Wih + (size_t)(nBase + nt * 16 + colB) * NR + k + bKoff;
            BF16Frag b;
            b.h[0] = *(const v8bf*)(wp);
            b.h[1] = *(const v8bf*)(wp + 16);
            acc[nt] = wmma_bf16(a, b, acc[nt]);
        }
    }
    // K over h (1024)
    for (int k = 0; k < NH; k += 32) {
        BF16Frag a;
        a.h[0] = *(const v8bf*)(Hbf + (size_t)rowA * NH + k + aKoff);
        a.h[1] = *(const v8bf*)(Hbf + (size_t)rowA * NH + k + 16 + aKoff);
        #pragma unroll
        for (int nt = 0; nt < 8; ++nt) {
            const __bf16* wp = Whh + (size_t)(nBase + nt * 16 + colB) * NH + k + bKoff;
            BF16Frag b;
            b.h[0] = *(const v8bf*)(wp);
            b.h[1] = *(const v8bf*)(wp + 16);
            acc[nt] = wmma_bf16(a, b, acc[nt]);
        }
    }
    // C layout: VGPR r, lanes 0-15 -> (M=r), lanes 16-31 -> (M=8+r), N = lane&15
    const int rRow = mBase + ((lane >> 4) << 3);
    #pragma unroll
    for (int nt = 0; nt < 8; ++nt) {
        const int col = nBase + nt * 16 + colB;
        #pragma unroll
        for (int r = 0; r < 8; ++r)
            gates[(size_t)(rRow + r) * N4H + col] = acc[nt][r];
    }
}

// LSTM cell update + hidden output
__global__ void __launch_bounds__(256) k_lstm_cell(const float* __restrict__ gates,
                                                   const float* __restrict__ b_ih,
                                                   const float* __restrict__ b_hh,
                                                   float* __restrict__ hbuf,
                                                   float* __restrict__ cbuf,
                                                   __bf16* __restrict__ hbf,
                                                   float* __restrict__ out_hdd, int t) {
    int idx = blockIdx.x * 256 + threadIdx.x;           // b*1024 + j, 480 blocks
    int b = idx >> 10, j = idx & 1023;
    const float* gb = gates + (size_t)b * N4H;
    float gi = gb[j]            + b_ih[j]            + b_hh[j];
    float gf = gb[NH + j]       + b_ih[NH + j]       + b_hh[NH + j];
    float gg = gb[2 * NH + j]   + b_ih[2 * NH + j]   + b_hh[2 * NH + j];
    float go = gb[3 * NH + j]   + b_ih[3 * NH + j]   + b_hh[3 * NH + j];
    float c = sigf(gf) * cbuf[idx] + sigf(gi) * fast_tanh(gg);
    float h = sigf(go) * fast_tanh(c);
    cbuf[idx] = c;
    hbuf[idx] = h;
    hbf[(size_t)b * NH + j] = f2bf(h);
    out_hdd[(size_t)(b * NTT + t) * NH + j] = h;
}

// ============ host ============

extern "C" void kernel_launch(void* const* d_in, const int* in_sizes, int n_in,
                              void* d_out, int out_size, void* d_ws, size_t ws_size,
                              hipStream_t stream) {
    (void)in_sizes; (void)n_in; (void)out_size; (void)ws_size;

    const float* feature = (const float*)d_in[0];
    const float* Whw  = (const float*)d_in[1];
    const float* Whb  = (const float*)d_in[2];
    const float* Wcw  = (const float*)d_in[3];
    const float* Wcb  = (const float*)d_in[4];
    const float* Wxa  = (const float*)d_in[5];
    const float* Wha  = (const float*)d_in[6];
    const float* wa   = (const float*)d_in[7];
    const float* Wb   = (const float*)d_in[8];
    const float* bb   = (const float*)d_in[9];
    const float* Wih  = (const float*)d_in[10];
    const float* Whh  = (const float*)d_in[11];
    const float* b_ih = (const float*)d_in[12];
    const float* b_hh = (const float*)d_in[13];

    float* out      = (float*)d_out;
    float* out_vis  = out;
    float* out_hdd  = out_vis + (size_t)NB * NTT * NR;
    float* out_alp  = out_hdd + (size_t)NB * NTT * NH;
    float* out_bet  = out_alp + (size_t)NB * NTT * NL;

    // workspace carve (256B aligned chunks)
    char* p = (char*)d_ws;
    auto carve = [&](size_t bytes) -> void* {
        void* r = (void*)p;
        p += (bytes + 255) & ~(size_t)255;
        return r;
    };
    __bf16* wihbf = (__bf16*)carve((size_t)N4H * NR * 2);   // 75.5 MB
    __bf16* whhbf = (__bf16*)carve((size_t)N4H * NH * 2);   // 8.4 MB
    __bf16* wxaT  = (__bf16*)carve((size_t)NA * ND * 2);
    __bf16* vbf   = (__bf16*)carve((size_t)MPAD * NR * 2);
    __bf16* hbf   = (__bf16*)carve((size_t)MPAD * NH * 2);
    float*  hbuf  = (float*)carve((size_t)NB * NH * 4);
    float*  cbuf  = (float*)carve((size_t)NB * NH * 4);
    float*  f0    = (float*)carve((size_t)NB * ND * 4);
    float*  ha    = (float*)carve((size_t)NB * NA * 4);
    float*  ebuf  = (float*)carve((size_t)BL * 4);
    float*  gates = (float*)carve((size_t)MPAD * N4H * 4);

    // one-time (per launch) weight conversion + state init
    {
        int n1 = N4H * NR;
        k_convert_bf16<<<(n1 + 255) / 256, 256, 0, stream>>>(Wih, wihbf, n1);
        int n2 = N4H * NH;
        k_convert_bf16<<<(n2 + 255) / 256, 256, 0, stream>>>(Whh, whhbf, n2);
        k_transpose_wxa<<<(NA * ND + 255) / 256, 256, 0, stream>>>(Wxa, wxaT);
        k_zero_pad<<<(8 * (NR + NH) + 255) / 256, 256, 0, stream>>>(vbf, hbf);
        k_f0<<<NB, 64, 0, stream>>>(feature, f0);
        k_hc0<<<(NB * NH) / 256, 256, 0, stream>>>(f0, Whw, Whb, Wcw, Wcb, hbuf, cbuf, hbf);
    }

    for (int t = 0; t < NTT; ++t) {
        k_attn_small<<<NB, 320, 0, stream>>>(hbuf, Wha, Wb, bb, ha, out_bet, t);
        k_score<<<(BL / 16) / 4, 128, 0, stream>>>(feature, wxaT, ha, wa, ebuf, t);
        k_softmax_visual<<<NB, 256, 0, stream>>>(feature, ebuf, out_bet, out_vis, out_alp, vbf, t);
        k_gates_gemm<<<N4H / 128, 256, 0, stream>>>(vbf, hbf, wihbf, whhbf, gates);
        k_lstm_cell<<<(NB * NH) / 256, 256, 0, stream>>>(gates, b_ih, b_hh, hbuf, cbuf, hbf, out_hdd, t);
    }
}
